// NSFMarginal_66030827209328
// MI455X (gfx1250) — matrix operand order repaired and checked by reading, not compiled, not run on hardware
//
#include <hip/hip_runtime.h>

// ---------------------------------------------------------------------------
// NSF marginal flow: fused 4-stage MLP (bf16 WMMA, f32 accum) + RQS spline.
// Workspace layout (d_ws), ~5.12 MB total:
//   [0)        w1t : bf16 [1024][256]   (W1 transposed)
//   [512KB)    w2t : bf16 [1024][1024]
//   [2.5MB)    w3t : bf16 [1024][1024]
//   [4.5MB)    wot : bf16 [192][1024]   (Wo transposed, rows 190..191 zero)
//   [~4.88MB)  part: f32  [1024]        per-block logdet partials
// ---------------------------------------------------------------------------

typedef unsigned short u16;
typedef __attribute__((ext_vector_type(16))) __bf16 v16bf;
typedef __attribute__((ext_vector_type(8)))  float  v8f;

#define NBINS 32
#define PPL   95           // params per spline layer (3*32-1)
#define TAILB 4.0f
#define ROWS  64           // rows per block
#define HDIM  1024
#define DDIM  256
#define NPAR  192          // padded real param width (real 190)
// LDS pitches padded to break bank-conflict alignment (stride mod 64 dwords != 0)
#define HP    1032         // bf16 activation pitch: 516 dwords -> rotate 4 banks/row
#define CP    264          // bf16 context pitch:   132 dwords -> rotate 4 banks/row
#define PP    193          // f32 param pitch: 193 dwords -> conflict-free scalar reads

__device__ __forceinline__ u16 f2bf(float f) {
  unsigned u = __float_as_uint(f);
  unsigned r = (u + 0x7FFFu + ((u >> 16) & 1u)) >> 16;   // round-to-nearest-even
  return (u16)r;
}

// ---------------------------------------------------------------------------
// Weight transpose + f32 -> bf16 downconvert (one thread per max-size element)
// ---------------------------------------------------------------------------
extern "C" __global__ void convert_weights(
    const float* __restrict__ W1, const float* __restrict__ W2,
    const float* __restrict__ W3, const float* __restrict__ Wo,
    u16* __restrict__ w1t, u16* __restrict__ w2t,
    u16* __restrict__ w3t, u16* __restrict__ wot)
{
  const int i = blockIdx.x * blockDim.x + threadIdx.x;   // [0, 1M)
  {   // W2 / W3: [1024][1024] -> Wt[n][k]
    const int n = i >> 10, k = i & 1023;
    w2t[i] = f2bf(W2[(size_t)k * HDIM + n]);
    w3t[i] = f2bf(W3[(size_t)k * HDIM + n]);
  }
  if (i < HDIM * DDIM) {   // W1: [256][1024] -> Wt[n][256]
    const int n = i >> 8, k = i & 255;
    w1t[i] = f2bf(W1[(size_t)k * HDIM + n]);
  }
  if (i < NPAR * HDIM) {   // Wo: [1024][190] -> Wt[n][1024], pad rows = 0
    const int n = i >> 10, k = i & 1023;
    wot[i] = (n < 190) ? f2bf(Wo[(size_t)k * 190 + n]) : (u16)0;
  }
}

// ---------------------------------------------------------------------------
// One GEMM stage: C[64xN] = act(A[64xK] @ W[KxN] + b).
// A bf16 in LDS (padded pitch), W bf16 transposed (N-major) in global.
// Each wave owns one 16-row M-tile (mt = wave&3) and walks its N-tiles in
// chunks of CH, sharing one A-fragment LDS load across CH WMMAs per k-step.
// ---------------------------------------------------------------------------
template<int CH, bool PARAMS>
__device__ __forceinline__ void gemm_stage(
    const u16* __restrict__ ldsA, int apitch, int K,
    const u16* __restrict__ wt, const float* __restrict__ bias, int realN,
    u16* __restrict__ ldsOut, float* __restrict__ paramsOut, int nTiles)
{
  const int lane = threadIdx.x & 31;
  const int wave = threadIdx.x >> 5;
  const int l15  = lane & 15;
  const int hl   = lane >> 4;
  const int mt   = wave & 3;                 // M-tile fixed per wave
  const int nt0  = wave >> 2;                // 0 or 1; N-tiles stride 2
  const int ntCount = nTiles / 2;            // per-wave N-tile count

  // A fragment base per ISA 16-bit A layout: lanes 0-15 hold K {0..7,16..23},
  // lanes 16-31 hold K {8..15,24..31} (hl*8 shift), one row per l15.
  const u16* aRow = ldsA + (size_t)(mt * 16 + l15) * apitch + hl * 8;

  for (int g = 0; g < ntCount; g += CH) {
    v8f acc[CH];
    const u16* bBase[CH];
#pragma unroll
    for (int c = 0; c < CH; ++c) {
      acc[c] = (v8f){};
      const int nt = nt0 + 2 * (g + c);
      // B fragment: lane l15 = column, hl selects K {0..7,16..23} / {8..15,24..31}
      bBase[c] = wt + (size_t)(nt * 16 + l15) * K + hl * 8;
    }

    for (int k0 = 0; k0 < K; k0 += 32) {
      union { v16bf v; uint4 q[2]; } ua;
      ua.q[0] = *(const uint4*)(aRow + k0);
      ua.q[1] = *(const uint4*)(aRow + k0 + 16);
#pragma unroll
      for (int c = 0; c < CH; ++c) {
        __builtin_prefetch((const void*)(bBase[c] + k0 + 128), 0, 1);
        union { v16bf v; uint4 q[2]; } ub;
        ub.q[0] = *(const uint4*)(bBase[c] + k0);
        ub.q[1] = *(const uint4*)(bBase[c] + k0 + 16);
        acc[c] = __builtin_amdgcn_wmma_f32_16x16x32_bf16(
            false, ua.v, false, ub.v, (short)0, acc[c], false, false);
      }
    }

#pragma unroll
    for (int c = 0; c < CH; ++c) {
      const int nt = nt0 + 2 * (g + c);
      const int n  = nt * 16 + l15;
      const float bv = (n < realN) ? bias[n] : 0.0f;
#pragma unroll
      for (int r = 0; r < 8; ++r) {
        const int row = mt * 16 + hl * 8 + r;      // D layout: vgpr r -> M=r(+8)
        float v = acc[c][r] + bv;
        if (PARAMS) {
          paramsOut[row * PP + n] = v;
        } else {
          v = fmaxf(v, 0.0f);                      // ReLU on hidden layers
          ldsOut[(size_t)row * HP + n] = f2bf(v);
        }
      }
    }
  }
}

// ---------------------------------------------------------------------------
// Rational-quadratic spline, forward, one element, one layer (K=32 bins)
// ---------------------------------------------------------------------------
__device__ __forceinline__ float softplusf_(float u) {
  return (u > 20.0f) ? u : log1pf(__expf(u));
}

__device__ float rqs_layer(const float* __restrict__ p, float xin, float& ldAcc) {
  const bool  inside = (xin >= -TAILB) && (xin <= TAILB);
  const float xc = fminf(fmaxf(xin, -TAILB), TAILB);

  float mw = -3.4e38f, mh = -3.4e38f;
  for (int i = 0; i < NBINS; ++i) {
    mw = fmaxf(mw, p[i]);
    mh = fmaxf(mh, p[NBINS + i]);
  }
  float ew[NBINS], eh[NBINS];
  float sw = 0.0f, sh = 0.0f;
  for (int i = 0; i < NBINS; ++i) {
    ew[i] = __expf(p[i] - mw);          sw += ew[i];
    eh[i] = __expf(p[NBINS + i] - mh);  sh += eh[i];
  }
  const float aw = (1.0f - 0.001f * NBINS) / sw;
  const float ah = (1.0f - 0.001f * NBINS) / sh;

  float cw[NBINS + 1], ch[NBINS + 1];
  cw[0] = -TAILB; ch[0] = -TAILB;
  float accw = 0.0f, acch = 0.0f;
  for (int i = 0; i < NBINS; ++i) {
    accw += 0.001f + aw * ew[i];
    acch += 0.001f + ah * eh[i];
    cw[i + 1] = 2.0f * TAILB * accw - TAILB;
    ch[i + 1] = 2.0f * TAILB * acch - TAILB;
  }
  cw[NBINS] = TAILB; ch[NBINS] = TAILB;

  int idx = 0;
  for (int i = 1; i <= NBINS; ++i)
    if (xc >= cw[i]) idx = i;
  if (idx > NBINS - 1) idx = NBINS - 1;

  // derivative pads are exactly 1.0 (MIN_D + softplus(log(e^{1-MIN_D}-1)))
  const float d0 = (idx == 0)         ? 1.0f : 0.001f + softplusf_(p[2 * NBINS + idx - 1]);
  const float d1 = (idx == NBINS - 1) ? 1.0f : 0.001f + softplusf_(p[2 * NBINS + idx]);

  const float wb    = cw[idx + 1] - cw[idx];
  const float hb    = ch[idx + 1] - ch[idx];
  const float delta = hb / wb;
  const float theta = (xc - cw[idx]) / wb;
  const float t1m   = theta * (1.0f - theta);
  const float omt   = 1.0f - theta;
  const float num   = hb * (delta * theta * theta + d0 * t1m);
  const float den   = delta + (d0 + d1 - 2.0f * delta) * t1m;
  const float dnum  = delta * delta *
                      (d1 * theta * theta + 2.0f * delta * t1m + d0 * omt * omt);
  float y = ch[idx] + num / den;
  float l = __logf(dnum) - 2.0f * __logf(den);
  if (!inside) { y = xin; l = 0.0f; }
  ldAcc += l;
  return y;
}

// ---------------------------------------------------------------------------
// Fused MLP + spline. 1024 blocks x 256 threads (8 waves); ~258 KB dynamic LDS.
// ---------------------------------------------------------------------------
extern "C" __global__ __launch_bounds__(256)
void nsf_fused(const float* __restrict__ context, const float* __restrict__ x,
               const float* __restrict__ b1, const float* __restrict__ b2,
               const float* __restrict__ b3, const float* __restrict__ bo,
               const u16* __restrict__ w1t, const u16* __restrict__ w2t,
               const u16* __restrict__ w3t, const u16* __restrict__ wot,
               float* __restrict__ out, float* __restrict__ part)
{
  extern __shared__ __align__(16) u16 smem[];
  u16*   bufA    = smem;                     // 64*1032 bf16 = 129 KB
  u16*   bufB    = smem + ROWS * HP;         // 64*1032 bf16 = 129 KB
  float* paramsF = (float*)smem;             // aliases bufA (free after stage 3)

  const int tid     = threadIdx.x;
  const int rowBase = blockIdx.x * ROWS;

  // Load context tile -> bf16 in bufA (padded pitch CP)
  for (int i = tid; i < ROWS * DDIM; i += blockDim.x) {
    const int r = i >> 8, k = i & 255;
    bufA[r * CP + k] = f2bf(context[(size_t)(rowBase + r) * DDIM + k]);
  }
  __syncthreads();

  gemm_stage<4,false>(bufA, CP, DDIM, w1t, b1, HDIM, bufB, nullptr, HDIM / 16);
  __syncthreads();
  gemm_stage<4,false>(bufB, HP, HDIM, w2t, b2, HDIM, bufA, nullptr, HDIM / 16);
  __syncthreads();
  gemm_stage<4,false>(bufA, HP, HDIM, w3t, b3, HDIM, bufB, nullptr, HDIM / 16);
  __syncthreads();
  gemm_stage<2,true >(bufB, HP, HDIM, wot, bo, 190, nullptr, paramsF, NPAR / 16);
  __syncthreads();

  // Spline tail: one thread per row
  float ld = 0.0f;
  if (tid < ROWS) {
    const float* pr = paramsF + tid * PP;
    float xv = x[rowBase + tid];
    xv = rqs_layer(pr,       xv, ld);
    xv = rqs_layer(pr + PPL, xv, ld);
    out[rowBase + tid] = xv;
  }

  // Deterministic block-level logdet reduction (bufB region is free now)
  float* red = (float*)(smem + ROWS * HP);
  if (tid < ROWS) red[tid] = ld;
  __syncthreads();
  if (tid == 0) {
    float s = 0.0f;
#pragma unroll
    for (int i = 0; i < ROWS; ++i) s += red[i];
    part[blockIdx.x] = s;
  }
}

// 16 blocks per batch (1024 rows / 64) -> deterministic 16-way sum per batch
extern "C" __global__ void nsf_reduce(const float* __restrict__ part,
                                      float* __restrict__ outld)
{
  const int b = threadIdx.x;
  if (b < 64) {
    float s = 0.0f;
#pragma unroll
    for (int j = 0; j < 16; ++j) s += part[b * 16 + j];
    outld[b] = s;
  }
}

// ---------------------------------------------------------------------------
extern "C" void kernel_launch(void* const* d_in, const int* in_sizes, int n_in,
                              void* d_out, int out_size, void* d_ws, size_t ws_size,
                              hipStream_t stream)
{
  const float* context = (const float*)d_in[0];
  const float* x  = (const float*)d_in[1];
  const float* W1 = (const float*)d_in[2];
  const float* b1 = (const float*)d_in[3];
  const float* W2 = (const float*)d_in[4];
  const float* b2 = (const float*)d_in[5];
  const float* W3 = (const float*)d_in[6];
  const float* b3 = (const float*)d_in[7];
  const float* Wo = (const float*)d_in[8];
  const float* bo = (const float*)d_in[9];
  float* out = (float*)d_out;

  u16* w1t = (u16*)d_ws;                         // 1024*256
  u16* w2t = w1t + (size_t)HDIM * DDIM;          // 1024*1024
  u16* w3t = w2t + (size_t)HDIM * HDIM;          // 1024*1024
  u16* wot = w3t + (size_t)HDIM * HDIM;          // 192*1024
  float* part = (float*)(wot + (size_t)NPAR * HDIM);  // 1024 floats

  // 1) weights -> transposed bf16 (once per launch, ~2.5M elements)
  convert_weights<<<4096, 256, 0, stream>>>(W1, W2, W3, Wo, w1t, w2t, w3t, wot);

  // 2) fused MLP + spline: 65536 rows / 64 per block, padded LDS
  const size_t ldsBytes = (size_t)2 * ROWS * HP * sizeof(u16);  // 264192 B
  nsf_fused<<<1024, 256, ldsBytes, stream>>>(
      context, x, b1, b2, b3, bo, w1t, w2t, w3t, wot, out, part);

  // 3) deterministic logdet finalize into out[65536..65599]
  nsf_reduce<<<1, 64, 0, stream>>>(part, out + 64 * 1024);
}
